// LearnedUpsampling_76940044140962
// MI455X (gfx1250) — matrix operand order repaired and checked by reading, not compiled, not run on hardware
//
#include <hip/hip_runtime.h>
#include <math.h>

#define B_  16
#define K_  256
#define R_  256
#define T_  1000
#define TT  63   // ceil(T_/16)

typedef __attribute__((ext_vector_type(2)))  float  v2f;
typedef __attribute__((ext_vector_type(8)))  float  v8f;
typedef __attribute__((ext_vector_type(16))) __bf16 v16bf;

union BFPack { unsigned int u[8]; v16bf v; };

__device__ __forceinline__ float fast_rcp(float x) {
    return __builtin_amdgcn_rcpf(x);           // v_rcp_f32 (TRANS, co-executes)
}
__device__ __forceinline__ float fast_silu(float x) {
    return x * fast_rcp(1.0f + __expf(-x));    // v_exp + v_rcp, no IEEE div
}
__device__ __forceinline__ unsigned int pack2bf(float a, float b) {
    unsigned int ua = __float_as_uint(a);
    ua += 0x7fffu + ((ua >> 16) & 1u);
    unsigned int ub = __float_as_uint(b);
    ub += 0x7fffu + ((ub >> 16) & 1u);
    return (ua >> 16) | (ub & 0xffff0000u);
}

// ---------------------------------------------------------------------------
// Kernel 0: duration scan -> s_k, e_k, pred_mel_len, pred_mel_mask
// ---------------------------------------------------------------------------
__global__ __launch_bounds__(256) void k0_scan(const float* __restrict__ dur,
                                               float* __restrict__ ws_sk,
                                               float* __restrict__ ws_ek,
                                               int*   __restrict__ ws_len,
                                               float* __restrict__ out_mask,
                                               float* __restrict__ out_len) {
    __shared__ float sc[K_];
    int b = blockIdx.x, tid = threadIdx.x;
    float d = dur[b * K_ + tid];
    sc[tid] = d;
    __syncthreads();
    for (int off = 1; off < K_; off <<= 1) {
        float add = (tid >= off) ? sc[tid - off] : 0.0f;
        __syncthreads();
        sc[tid] += add;
        __syncthreads();
    }
    float e = sc[tid];
    ws_ek[b * K_ + tid] = e;
    ws_sk[b * K_ + tid] = e - d;
    float total = sc[K_ - 1];
    int len = (int)rintf(total);
    if (len > T_) len = T_;
    if (len < 0) len = 0;
    if (tid == 0) { ws_len[b] = len; out_len[b] = (float)len; }
    for (int t = tid; t < T_; t += 256)
        out_mask[b * T_ + t] = (t >= len) ? 1.0f : 0.0f;
}

// ---------------------------------------------------------------------------
// Kernel 1: both conv blocks (conv + BN-scale + silu + LN over 8 channels)
// then pre-fold base_{w,c} = y(8) @ w1[2:] + b1  -> (B,K,16) each
// ---------------------------------------------------------------------------
__global__ __launch_bounds__(256) void k1_conv(
    const float* __restrict__ V,
    const float* __restrict__ ccw, const float* __restrict__ ccb,
    const float* __restrict__ bncg, const float* __restrict__ bncb,
    const float* __restrict__ lncg, const float* __restrict__ lncb,
    const float* __restrict__ cww, const float* __restrict__ cwb,
    const float* __restrict__ bnwg, const float* __restrict__ bnwb,
    const float* __restrict__ lnwg, const float* __restrict__ lnwb,
    const float* __restrict__ swcw1, const float* __restrict__ swcb1,
    const float* __restrict__ swww1, const float* __restrict__ swwb1,
    float* __restrict__ basew, float* __restrict__ basec)
{
    __shared__ float wc[3 * 256 * 8];
    __shared__ float ww[3 * 256 * 8];
    int b = blockIdx.x, k = threadIdx.x;
    for (int i = k; i < 3 * 256 * 8; i += 256) { wc[i] = ccw[i]; ww[i] = cww[i]; }
    __syncthreads();

    float accw[8], accc[8];
    #pragma unroll
    for (int c = 0; c < 8; ++c) { accw[c] = cwb[c]; accc[c] = ccb[c]; }

    for (int dk = 0; dk < 3; ++dk) {
        int kk = k + dk - 1;
        if (kk < 0 || kk >= K_) continue;
        const float* vrow = V + ((size_t)b * K_ + kk) * R_;
        for (int r = 0; r < R_; ++r) {
            float v = vrow[r];
            const float* pw = &ww[(dk * 256 + r) * 8];
            const float* pc = &wc[(dk * 256 + r) * 8];
            #pragma unroll
            for (int c = 0; c < 8; ++c) { accw[c] += v * pw[c]; accc[c] += v * pc[c]; }
        }
    }

    const float inv_bn = rsqrtf(1.0f + 1e-5f);
    float yw[8], yc[8];
    // w branch: BN-scale, silu, LN(8)
    float mu = 0.0f;
    #pragma unroll
    for (int c = 0; c < 8; ++c) { yw[c] = fast_silu(accw[c] * (bnwg[c] * inv_bn) + bnwb[c]); mu += yw[c]; }
    mu *= 0.125f;
    float var = 0.0f;
    #pragma unroll
    for (int c = 0; c < 8; ++c) { float d = yw[c] - mu; var += d * d; }
    var *= 0.125f;
    float rs = rsqrtf(var + 1e-5f);
    #pragma unroll
    for (int c = 0; c < 8; ++c) yw[c] = (yw[c] - mu) * rs * lnwg[c] + lnwb[c];
    // c branch
    mu = 0.0f;
    #pragma unroll
    for (int c = 0; c < 8; ++c) { yc[c] = fast_silu(accc[c] * (bncg[c] * inv_bn) + bncb[c]); mu += yc[c]; }
    mu *= 0.125f;
    var = 0.0f;
    #pragma unroll
    for (int c = 0; c < 8; ++c) { float d = yc[c] - mu; var += d * d; }
    var *= 0.125f;
    rs = rsqrtf(var + 1e-5f);
    #pragma unroll
    for (int c = 0; c < 8; ++c) yc[c] = (yc[c] - mu) * rs * lncg[c] + lncb[c];

    size_t ob = ((size_t)(b * K_ + k)) * 16;
    #pragma unroll
    for (int j = 0; j < 16; ++j) {
        float bw = swwb1[j], bc = swcb1[j];
        #pragma unroll
        for (int c = 0; c < 8; ++c) {
            bw += yw[c] * swww1[(2 + c) * 16 + j];
            bc += yc[c] * swcw1[(2 + c) * 16 + j];
        }
        basew[ob + j] = bw;
        basec[ob + j] = bc;
    }
}

// ---------------------------------------------------------------------------
// Kernel 1b: V (f32, row-major) -> bf16 packed in WMMA-B (32x16) lane layout
// index: [b][kc(8)][nt(16)][j(8)][lane(32)]  (u32 holding 2 bf16)
// lane<16:  K = kc*32 + 2j   + {0,1}, N = nt*16 + (lane&15)
// lane>=16: K = kc*32 + 16+2j + {0,1}
// ---------------------------------------------------------------------------
__global__ __launch_bounds__(256) void k1b_vbf(const float* __restrict__ V,
                                               unsigned int* __restrict__ vbf) {
    int id = blockIdx.x;                 // b*128 + kc*16 + nt
    int b = id >> 7, kc = (id >> 4) & 7, nt = id & 15;
    int tid = threadIdx.x;
    int j = tid >> 5, lane = tid & 31, half = lane >> 4, l15 = lane & 15;
    int kbase = kc * 32 + half * 16 + 2 * j;
    int n = nt * 16 + l15;
    float f0 = V[((size_t)b * K_ + kbase) * R_ + n];
    float f1 = V[((size_t)b * K_ + kbase + 1) * R_ + n];
    vbf[((((size_t)b * 8 + kc) * 16 + nt) * 8 + j) * 32 + lane] = pack2bf(f0, f1);
}

// ---------------------------------------------------------------------------
// Kernel 2: fused per (b, 16-row t-tile):
//   async global->LDS staging of base_w/base_c (ASYNCcnt path)
//   pass A: swish(w) -> logits via f32 WMMA ; softmax over k
//   pass B: swish(c) -> agg accumulation (LDS atomics)
//   W@V (bf16 WMMA 16x16x32) + agg@lin_e (f32 WMMA) + bias + LN(R) + mask
// ---------------------------------------------------------------------------
__global__ __launch_bounds__(256) void k2_main(
    const unsigned char* __restrict__ tmask,
    const float* __restrict__ ws_sk, const float* __restrict__ ws_ek,
    const int*   __restrict__ ws_len,
    const float* __restrict__ basew_g, const float* __restrict__ basec_g,
    const float* __restrict__ swcw1, const float* __restrict__ swcw2, const float* __restrict__ swcb2,
    const float* __restrict__ swww1, const float* __restrict__ swww2, const float* __restrict__ swwb2,
    const float* __restrict__ projw, const float* __restrict__ projb,
    const unsigned int* __restrict__ vbf,
    const float* __restrict__ linew, const float* __restrict__ lineb,
    const float* __restrict__ lnog, const float* __restrict__ lnob,
    float* __restrict__ out_up, float* __restrict__ out_W)
{
    __shared__ float lds_sk[K_], lds_ek[K_];
    __shared__ float lds_bw[K_ * 16], lds_bc[K_ * 16];
    __shared__ float lds_W[16 * K_];
    __shared__ float lds_agg[16 * 16];
    __shared__ float lds_rs[16], lds_rss[16];
    __shared__ unsigned char lds_tm[K_];

    int bi = blockIdx.x;
    int b = bi / TT, tile = bi % TT;
    int t0 = tile * 16;
    int tid = threadIdx.x;
    int wave = tid >> 5, lane = tid & 31, half = lane >> 4, l15 = lane & 15;

    // ---- async DMA of base_w / base_c straight into LDS (no VGPR staging) ----
    {
        unsigned ldsw = (unsigned)(size_t)(&lds_bw[0]);
        unsigned ldsc = (unsigned)(size_t)(&lds_bc[0]);
        const float* gw = basew_g + (size_t)b * K_ * 16;
        const float* gc0 = basec_g + (size_t)b * K_ * 16;
        #pragma unroll
        for (int it = 0; it < 4; ++it) {               // 4 * 256thr * 16B = 16KB each
            unsigned goff = (unsigned)tid * 16u + (unsigned)it * 4096u;
            unsigned dW = ldsw + goff;
            unsigned dC = ldsc + goff;
            asm volatile("global_load_async_to_lds_b128 %0, %1, %2 offset:0"
                         :: "v"(dW), "v"(goff), "s"(gw) : "memory");
            asm volatile("global_load_async_to_lds_b128 %0, %1, %2 offset:0"
                         :: "v"(dC), "v"(goff), "s"(gc0) : "memory");
        }
    }

    for (int i = tid; i < K_; i += 256) {
        lds_sk[i] = ws_sk[b * K_ + i];
        lds_ek[i] = ws_ek[b * K_ + i];
        lds_tm[i] = tmask[b * K_ + i];
    }
    lds_agg[tid & 255] = 0.0f;
    if (tid < 16) { lds_rs[tid] = 0.0f; lds_rss[tid] = 0.0f; }
    int len = ws_len[b];

    // per-lane swish weights: lane owns 8 features (A-operand f32 layout order)
    float w1w0[8], w1w1[8], w1c0[8], w1c1[8];
    #pragma unroll
    for (int q = 0; q < 8; ++q) {
        int f = ((q >> 1) << 2) + half * 2 + (q & 1);
        w1w0[q] = swww1[f];       w1w1[q] = swww1[16 + f];
        w1c0[q] = swcw1[f];       w1c1[q] = swcw1[16 + f];
    }
    v2f bw2[4], bc2[4];
    #pragma unroll
    for (int c = 0; c < 4; ++c) {
        int p0 = 4 * c + 2 * half;
        bw2[c] = (v2f){ swww2[p0 * 16 + l15], swww2[(p0 + 1) * 16 + l15] };
        bc2[c] = (v2f){ swcw2[p0 * 16 + l15], swcw2[(p0 + 1) * 16 + l15] };
    }
    float b2w = swwb2[l15], b2c = swcb2[l15];
    float pj = projw[l15], pjb = projb[0];

    int trow = t0 + l15;               // A-matrix row (M = lane&15, both halves)
    float tval = (float)(trow + 1);
    bool melm = (trow >= len);
    const float NEG_INF = __int_as_float(0xff800000);

    // wait for our async LDS writes before the workgroup barrier
    asm volatile("s_wait_asynccnt 0x0" ::: "memory");
    __syncthreads();

    // ---------------- pass A: logits (w branch) ----------------
    for (int i = 0; i < 32; ++i) {
        int k = wave * 32 + i;
        bool tm = lds_tm[k] != 0;
        bool am = tm || melm;
        float S = am ? 0.0f : (tval - lds_sk[k]);
        float E = am ? 0.0f : (lds_ek[k] - tval);
        float h[8];
        #pragma unroll
        for (int q = 0; q < 8; ++q) {
            int f = ((q >> 1) << 2) + half * 2 + (q & 1);
            h[q] = fast_silu(S * w1w0[q] + E * w1w1[q] + lds_bw[k * 16 + f]);
        }
        v8f acc = {};
        #pragma unroll
        for (int c = 0; c < 4; ++c) {
            v2f a = (v2f){ h[2 * c], h[2 * c + 1] };
            acc = __builtin_amdgcn_wmma_f32_16x16x4_f32(false, a, false, bw2[c],
                                                        (short)0, acc, false, false);
        }
        float lg[8];
        #pragma unroll
        for (int r = 0; r < 8; ++r) {
            float p = fast_silu(acc[r] + b2w) * pj;
            p += __shfl_xor(p, 1);
            p += __shfl_xor(p, 2);
            p += __shfl_xor(p, 4);
            p += __shfl_xor(p, 8);
            lg[r] = p + pjb;
        }
        if (l15 == 0) {
            #pragma unroll
            for (int r = 0; r < 8; ++r)
                lds_W[(half * 8 + r) * K_ + k] = tm ? NEG_INF : lg[r];
        }
    }
    __syncthreads();

    // ---------------- softmax over k (2 rows per wave) ----------------
    {
        int row = wave * 2 + half;
        float m = NEG_INF;
        for (int i = 0; i < 16; ++i)
            m = fmaxf(m, lds_W[row * K_ + l15 + 16 * i]);
        m = fmaxf(m, __shfl_xor(m, 1));
        m = fmaxf(m, __shfl_xor(m, 2));
        m = fmaxf(m, __shfl_xor(m, 4));
        m = fmaxf(m, __shfl_xor(m, 8));
        float s = 0.0f;
        for (int i = 0; i < 16; ++i) {
            int c = l15 + 16 * i;
            float e = __expf(lds_W[row * K_ + c] - m);
            lds_W[row * K_ + c] = e;
            s += e;
        }
        s += __shfl_xor(s, 1);
        s += __shfl_xor(s, 2);
        s += __shfl_xor(s, 4);
        s += __shfl_xor(s, 8);
        float invs = fast_rcp(s);
        bool rm = (t0 + row) >= len;
        for (int i = 0; i < 16; ++i) {
            int c = l15 + 16 * i;
            lds_W[row * K_ + c] = rm ? 0.0f : lds_W[row * K_ + c] * invs;
        }
    }
    __syncthreads();

    // write W output
    for (int i = tid; i < 16 * K_; i += 256) {
        int row = i >> 8;
        int t = t0 + row;
        if (t < T_) out_W[((size_t)b * T_ + t) * K_ + (i & 255)] = lds_W[i];
    }

    // ---------------- pass B: agg (c branch) ----------------
    float aggacc[8] = {0.f, 0.f, 0.f, 0.f, 0.f, 0.f, 0.f, 0.f};
    for (int i = 0; i < 32; ++i) {
        int k = wave * 32 + i;
        bool am = (lds_tm[k] != 0) || melm;
        float S = am ? 0.0f : (tval - lds_sk[k]);
        float E = am ? 0.0f : (lds_ek[k] - tval);
        float h[8];
        #pragma unroll
        for (int q = 0; q < 8; ++q) {
            int f = ((q >> 1) << 2) + half * 2 + (q & 1);
            h[q] = fast_silu(S * w1c0[q] + E * w1c1[q] + lds_bc[k * 16 + f]);
        }
        v8f acc = {};
        #pragma unroll
        for (int c = 0; c < 4; ++c) {
            v2f a = (v2f){ h[2 * c], h[2 * c + 1] };
            acc = __builtin_amdgcn_wmma_f32_16x16x4_f32(false, a, false, bc2[c],
                                                        (short)0, acc, false, false);
        }
        #pragma unroll
        for (int r = 0; r < 8; ++r)
            aggacc[r] += lds_W[(half * 8 + r) * K_ + k] * fast_silu(acc[r] + b2c);
    }
    #pragma unroll
    for (int r = 0; r < 8; ++r)
        atomicAdd(&lds_agg[(half * 8 + r) * 16 + l15], aggacc[r]);
    __syncthreads();

    // ---------------- W (bf16 A-layout, 8 K-chunks of 32) ----------------
    BFPack aW[8];
    #pragma unroll
    for (int kc = 0; kc < 8; ++kc) {
        #pragma unroll
        for (int j = 0; j < 8; ++j) {
            int k0 = kc * 32 + half * 16 + 2 * j;
            aW[kc].u[j] = pack2bf(lds_W[l15 * K_ + k0], lds_W[l15 * K_ + k0 + 1]);
        }
    }

    // ---------------- up = W@V + agg@lin_e + b ; LN(R) ----------------
    float x[2][8];
    float gc[2], bc_[2];
    #pragma unroll
    for (int w2i = 0; w2i < 2; ++w2i) {
        int nt = wave * 2 + w2i;
        int col = nt * 16 + l15;
        v8f acc = {};
        #pragma unroll
        for (int kc = 0; kc < 8; ++kc) {
            BFPack bv;
            const unsigned int* src =
                vbf + ((((size_t)b * 8 + kc) * 16 + nt) * 8) * 32 + lane;
            #pragma unroll
            for (int j = 0; j < 8; ++j) bv.u[j] = src[j * 32];
            acc = __builtin_amdgcn_wmma_f32_16x16x32_bf16(false, aW[kc].v, false, bv.v,
                                                          (short)0, acc, false, false);
        }
        #pragma unroll
        for (int c = 0; c < 4; ++c) {
            v2f a = (v2f){ lds_agg[l15 * 16 + 4 * c + 2 * half],
                           lds_agg[l15 * 16 + 4 * c + 2 * half + 1] };
            int p0 = 4 * c + 2 * half;
            v2f bm = (v2f){ linew[p0 * R_ + col], linew[(p0 + 1) * R_ + col] };
            acc = __builtin_amdgcn_wmma_f32_16x16x4_f32(false, a, false, bm,
                                                        (short)0, acc, false, false);
        }
        float leb = lineb[col];
        #pragma unroll
        for (int r = 0; r < 8; ++r) x[w2i][r] = acc[r] + leb;
        gc[w2i] = lnog[col];
        bc_[w2i] = lnob[col];
    }
    #pragma unroll
    for (int r = 0; r < 8; ++r) {
        float s1 = x[0][r] + x[1][r];
        float s2 = x[0][r] * x[0][r] + x[1][r] * x[1][r];
        s1 += __shfl_xor(s1, 1); s2 += __shfl_xor(s2, 1);
        s1 += __shfl_xor(s1, 2); s2 += __shfl_xor(s2, 2);
        s1 += __shfl_xor(s1, 4); s2 += __shfl_xor(s2, 4);
        s1 += __shfl_xor(s1, 8); s2 += __shfl_xor(s2, 8);
        if (l15 == 0) {
            atomicAdd(&lds_rs[half * 8 + r], s1);
            atomicAdd(&lds_rss[half * 8 + r], s2);
        }
    }
    __syncthreads();
    #pragma unroll
    for (int r = 0; r < 8; ++r) {
        int row = half * 8 + r;
        int t = t0 + row;
        float mu = lds_rs[row] * (1.0f / 256.0f);
        float var = lds_rss[row] * (1.0f / 256.0f) - mu * mu;
        float rstd = rsqrtf(var + 1e-5f);
        bool zero = (t >= len);
        if (t < T_) {
            #pragma unroll
            for (int w2i = 0; w2i < 2; ++w2i) {
                int col = (wave * 2 + w2i) * 16 + l15;
                float y = (x[w2i][r] - mu) * rstd * gc[w2i] + bc_[w2i];
                out_up[((size_t)b * T_ + t) * R_ + col] = zero ? 0.0f : y;
            }
        }
    }
}

// ---------------------------------------------------------------------------
extern "C" void kernel_launch(void* const* d_in, const int* in_sizes, int n_in,
                              void* d_out, int out_size, void* d_ws, size_t ws_size,
                              hipStream_t stream) {
    (void)in_sizes; (void)n_in; (void)out_size; (void)ws_size;
    const float* dur      = (const float*)d_in[0];
    const float* V        = (const float*)d_in[1];
    /* pos_text d_in[2] unused */
    const unsigned char* text_mask = (const unsigned char*)d_in[3];
    const float* conv_c_w = (const float*)d_in[4];
    const float* conv_c_b = (const float*)d_in[5];
    const float* bn_c_g   = (const float*)d_in[6];
    const float* bn_c_b   = (const float*)d_in[7];
    const float* ln_c_g   = (const float*)d_in[8];
    const float* ln_c_b   = (const float*)d_in[9];
    const float* conv_w_w = (const float*)d_in[10];
    const float* conv_w_b = (const float*)d_in[11];
    const float* bn_w_g   = (const float*)d_in[12];
    const float* bn_w_b   = (const float*)d_in[13];
    const float* ln_w_g   = (const float*)d_in[14];
    const float* ln_w_b   = (const float*)d_in[15];
    const float* swc_w1   = (const float*)d_in[16];
    const float* swc_b1   = (const float*)d_in[17];
    const float* swc_w2   = (const float*)d_in[18];
    const float* swc_b2   = (const float*)d_in[19];
    const float* sww_w1   = (const float*)d_in[20];
    const float* sww_b1   = (const float*)d_in[21];
    const float* sww_w2   = (const float*)d_in[22];
    const float* sww_b2   = (const float*)d_in[23];
    const float* proj_w_w = (const float*)d_in[24];
    const float* proj_w_b = (const float*)d_in[25];
    const float* lin_e_w  = (const float*)d_in[26];
    const float* lin_e_b  = (const float*)d_in[27];
    const float* ln_o_g   = (const float*)d_in[28];
    const float* ln_o_b   = (const float*)d_in[29];

    float* ws = (float*)d_ws;
    float* ws_sk    = ws;                       // 4096
    float* ws_ek    = ws + 4096;                // 4096
    int*   ws_len   = (int*)(ws + 8192);        // 16 (pad to 16 floats)
    float* ws_basew = ws + 8208;                // 65536
    float* ws_basec = ws + 8208 + 65536;        // 65536
    unsigned int* ws_vbf = (unsigned int*)(ws + 8208 + 2 * 65536); // 524288 u32

    float* out      = (float*)d_out;
    float* out_up   = out;                              // B*T*R
    float* out_mask = out + (size_t)B_ * T_ * R_;       // B*T
    float* out_len  = out_mask + (size_t)B_ * T_;       // B
    float* out_W    = out_len + B_;                     // B*T*K

    k0_scan<<<B_, 256, 0, stream>>>(dur, ws_sk, ws_ek, ws_len, out_mask, out_len);

    k1_conv<<<B_, 256, 0, stream>>>(V,
        conv_c_w, conv_c_b, bn_c_g, bn_c_b, ln_c_g, ln_c_b,
        conv_w_w, conv_w_b, bn_w_g, bn_w_b, ln_w_g, ln_w_b,
        swc_w1, swc_b1, sww_w1, sww_b1,
        ws_basew, ws_basec);

    k1b_vbf<<<B_ * 8 * 16, 256, 0, stream>>>(V, ws_vbf);

    k2_main<<<B_ * TT, 256, 0, stream>>>(text_mask,
        ws_sk, ws_ek, ws_len, ws_basew, ws_basec,
        swc_w1, swc_w2, swc_b2,
        sww_w1, sww_w2, sww_b2,
        proj_w_w, proj_w_b,
        ws_vbf, lin_e_w, lin_e_b, ln_o_g, ln_o_b,
        out_up, out_W);
}
